// MultiHeadAttention_48430051230465
// MI455X (gfx1250) — compile-verified, matrix operand order checked by reading, not run on hardware
//
#include <hip/hip_runtime.h>
#include <hip/hip_bf16.h>

typedef unsigned short ushort_t;
typedef unsigned int   uint_t;
typedef __attribute__((ext_vector_type(16))) __bf16 v16bf;
typedef __attribute__((ext_vector_type(8)))  float  v8f;
typedef __attribute__((ext_vector_type(4)))  uint_t v4u;

#define BATCH 128
#define SEQ   256
#define EMB   384
#define HEADS 6
#define HS    64
#define MROWS (BATCH*SEQ)   // 32768

union FragU { v16bf v; v4u q[2]; };
union PackU { v4u v; ushort_t s[8]; };

__device__ __forceinline__ ushort_t f2bf(float f) {
    uint_t u = __builtin_bit_cast(uint_t, f);
    u += 0x7FFFu + ((u >> 16) & 1u);      // round-to-nearest-even
    return (ushort_t)(u >> 16);
}

// ---------------- conversion kernels ----------------
__global__ __launch_bounds__(256) void k_cvt_x(const float* __restrict__ x,
                                               ushort_t* __restrict__ xb) {
    int i = blockIdx.x * 256 + threadIdx.x;   // MROWS*EMB elements, divides evenly
    xb[i] = f2bf(x[i]);
}

// wT layout: [4][N=384][K=384], order Wq,Wk,Wv,Wp, transposed (dst[n*384+k]=src[k*384+n])
__global__ __launch_bounds__(256) void k_cvt_w(const float* __restrict__ Wq,
                                               const float* __restrict__ Wk,
                                               const float* __restrict__ Wv,
                                               const float* __restrict__ Wp,
                                               ushort_t* __restrict__ wT) {
    int i = blockIdx.x * 256 + threadIdx.x;   // 4*384*384 elements
    int w   = i / (EMB*EMB);
    int rem = i % (EMB*EMB);
    int k = rem / EMB, n = rem % EMB;
    const float* src = (w == 0) ? Wq : (w == 1) ? Wk : (w == 2) ? Wv : Wp;
    wT[(size_t)w*EMB*EMB + (size_t)n*EMB + k] = f2bf(src[(size_t)k*EMB + n]);
}

// ---------------- fused QKV projection ----------------
// grid (MROWS/128, EMB/128, 3). 4 waves/block (2x2); each wave does a 64x64 tile
// (16 WMMAs per 8 fragment loads per K-step).
// z=0 -> Q [B,H,T,hs] ; z=1 -> K [B,H,T,hs] ; z=2 -> V^T [B,H,hs,T]
__global__ __launch_bounds__(128) void k_qkv_gemm(const ushort_t* __restrict__ xb,
                                                  const ushort_t* __restrict__ wT,
                                                  ushort_t* __restrict__ qb,
                                                  ushort_t* __restrict__ kb,
                                                  ushort_t* __restrict__ vTb) {
    const int z    = blockIdx.z;
    const ushort_t* w = wT + (size_t)z * EMB * EMB;
    const int wid  = threadIdx.x >> 5;
    const int lane = threadIdx.x & 31;
    const int n16  = lane & 15, half = lane >> 4;
    const int wm   = wid >> 1, wn = wid & 1;
    const int mrow = blockIdx.x * 128 + wm * 64;
    const int ncol = blockIdx.y * 128 + wn * 64;

    v8f c[4][4] = {};
    for (int k0 = 0; k0 < EMB; k0 += 32) {
        FragU a[4], bfr[4];
#pragma unroll
        for (int i = 0; i < 4; ++i) {
            const ushort_t* ap = xb + (size_t)(mrow + i*16 + n16) * EMB + k0 + 8*half;
            a[i].q[0] = *(const v4u*)(ap);
            a[i].q[1] = *(const v4u*)(ap + 16);
        }
#pragma unroll
        for (int j = 0; j < 4; ++j) {
            const ushort_t* bp = w + (size_t)(ncol + j*16 + n16) * EMB + k0 + 8*half;
            bfr[j].q[0] = *(const v4u*)(bp);
            bfr[j].q[1] = *(const v4u*)(bp + 16);
        }
#pragma unroll
        for (int i = 0; i < 4; ++i)
#pragma unroll
            for (int j = 0; j < 4; ++j)
                c[i][j] = __builtin_amdgcn_wmma_f32_16x16x32_bf16(
                    false, a[i].v, false, bfr[j].v, (short)0, c[i][j], false, false);
    }

#pragma unroll
    for (int i = 0; i < 4; ++i) {
#pragma unroll
        for (int j = 0; j < 4; ++j) {
            const int colb = ncol + j*16 + n16;
            const int h = colb >> 6, d = colb & 63;
            if (z == 2) {                       // V^T, packed 16B store (8 consecutive t)
                PackU pk;
#pragma unroll
                for (int r = 0; r < 8; ++r) pk.s[r] = f2bf(c[i][j][r]);
                const int rowb = mrow + i*16;
                const int bb = rowb >> 8, tb = rowb & 255;
                *(v4u*)(vTb + ((((size_t)bb*HEADS + h)*HS + d)*SEQ) + tb + 8*half) = pk.v;
            } else {
                ushort_t* dst = (z == 0) ? qb : kb;
#pragma unroll
                for (int r = 0; r < 8; ++r) {
                    const int row = mrow + i*16 + 8*half + r;
                    const int bb = row >> 8, t = row & 255;
                    dst[(((size_t)bb*HEADS + h)*SEQ + t)*HS + d] = f2bf(c[i][j][r]);
                }
            }
        }
    }
}

// ---------------- flash attention ----------------
// One wave per block; block = 16 queries of one (b,h). Causal, online softmax.
__global__ __launch_bounds__(32) void k_attn(const ushort_t* __restrict__ qb,
                                             const ushort_t* __restrict__ kb,
                                             const ushort_t* __restrict__ vTb,
                                             ushort_t* __restrict__ ab) {
    __shared__ ushort_t ptile[16 * 32];       // 1 KB P staging tile
    const int lane = threadIdx.x & 31;
    const int n16  = lane & 15, half = lane >> 4;
    const int bid  = blockIdx.x;
    const int qt = bid & 15;
    const int h  = (bid >> 4) % HEADS;
    const int b  = bid / (16 * HEADS);
    const int q0 = qt * 16;

    const ushort_t* qh = qb  + (((size_t)b*HEADS + h)*SEQ)*HS;
    const ushort_t* kh = kb  + (((size_t)b*HEADS + h)*SEQ)*HS;
    const ushort_t* vh = vTb + (((size_t)b*HEADS + h)*HS)*SEQ;

    FragU qa[2];
#pragma unroll
    for (int cc = 0; cc < 2; ++cc) {
        const ushort_t* p = qh + (size_t)(q0 + n16)*HS + 32*cc + 8*half;
        qa[cc].q[0] = *(const v4u*)(p);
        qa[cc].q[1] = *(const v4u*)(p + 16);
    }

    float mrun[8], lrun[8];
    v8f o[4] = {};
#pragma unroll
    for (int r = 0; r < 8; ++r) { mrun[r] = -1e30f; lrun[r] = 0.0f; }
    const float scale = 0.05103103630798287f;   // 384^-0.5 (reference uses C, not hs)

    const int send = q0 + 16;
    for (int s = 0; s < send; s += 32) {
        v8f s0 = {}, s1 = {};
#pragma unroll
        for (int cc = 0; cc < 2; ++cc) {
            FragU k0f, k1f;
            const ushort_t* p0 = kh + (size_t)(s + n16)*HS + 32*cc + 8*half;
            k0f.q[0] = *(const v4u*)(p0);  k0f.q[1] = *(const v4u*)(p0 + 16);
            const ushort_t* p1 = kh + (size_t)(s + 16 + n16)*HS + 32*cc + 8*half;
            k1f.q[0] = *(const v4u*)(p1);  k1f.q[1] = *(const v4u*)(p1 + 16);
            s0 = __builtin_amdgcn_wmma_f32_16x16x32_bf16(false, qa[cc].v, false, k0f.v, (short)0, s0, false, false);
            s1 = __builtin_amdgcn_wmma_f32_16x16x32_bf16(false, qa[cc].v, false, k1f.v, (short)0, s1, false, false);
        }

        float p0v[8], p1v[8], alpha[8];
        const int qbase = q0 + 8*half;
#pragma unroll
        for (int r = 0; r < 8; ++r) {
            const int qi = qbase + r;
            float x0 = s0[r] * scale; if (s + n16      > qi) x0 = -1e30f;
            float x1 = s1[r] * scale; if (s + 16 + n16 > qi) x1 = -1e30f;
            float tm = fmaxf(x0, x1);
#pragma unroll
            for (int mm = 8; mm >= 1; mm >>= 1) tm = fmaxf(tm, __shfl_xor(tm, mm, 32));
            const float mnew = fmaxf(mrun[r], tm);
            const float e0 = __expf(x0 - mnew);
            const float e1 = __expf(x1 - mnew);
            float ts = e0 + e1;
#pragma unroll
            for (int mm = 8; mm >= 1; mm >>= 1) ts += __shfl_xor(ts, mm, 32);
            alpha[r] = __expf(mrun[r] - mnew);
            lrun[r]  = lrun[r] * alpha[r] + ts;
            mrun[r]  = mnew;
            p0v[r] = e0; p1v[r] = e1;
        }
#pragma unroll
        for (int j = 0; j < 4; ++j)
#pragma unroll
            for (int r = 0; r < 8; ++r) o[j][r] *= alpha[r];

        __syncthreads();                        // WAR vs previous iteration reads
#pragma unroll
        for (int r = 0; r < 8; ++r) {
            const int row = r + 8*half;
            ptile[row*32 + n16]      = f2bf(p0v[r]);
            ptile[row*32 + 16 + n16] = f2bf(p1v[r]);
        }
        __syncthreads();                        // store -> load ordering

        FragU pa;
        const ushort_t* pp = ptile + n16*32 + 8*half;
        pa.q[0] = *(const v4u*)(pp);
        pa.q[1] = *(const v4u*)(pp + 16);
#pragma unroll
        for (int j = 0; j < 4; ++j) {
            FragU vf;
            const ushort_t* vp = vh + (size_t)(16*j + n16)*SEQ + s + 8*half;
            vf.q[0] = *(const v4u*)(vp);
            vf.q[1] = *(const v4u*)(vp + 16);
            o[j] = __builtin_amdgcn_wmma_f32_16x16x32_bf16(false, pa.v, false, vf.v, (short)0, o[j], false, false);
        }
    }

#pragma unroll
    for (int r = 0; r < 8; ++r) {
        const float inv = 1.0f / lrun[r];
        const int row = b*SEQ + q0 + r + 8*half;
#pragma unroll
        for (int j = 0; j < 4; ++j) {
            const int col = h*HS + 16*j + n16;
            ab[(size_t)row*EMB + col] = f2bf(o[j][r] * inv);
        }
    }
}

// ---------------- output projection + bias ----------------
// grid (MROWS/128, EMB/128). 4 waves/block (2x2); each wave a 64x64 tile.
__global__ __launch_bounds__(128) void k_proj_gemm(const ushort_t* __restrict__ ab,
                                                   const ushort_t* __restrict__ wpT,
                                                   const float* __restrict__ bp,
                                                   float* __restrict__ out) {
    const int wid  = threadIdx.x >> 5;
    const int lane = threadIdx.x & 31;
    const int n16  = lane & 15, half = lane >> 4;
    const int wm   = wid >> 1, wn = wid & 1;
    const int mrow = blockIdx.x * 128 + wm * 64;
    const int ncol = blockIdx.y * 128 + wn * 64;

    v8f c[4][4] = {};
    for (int k0 = 0; k0 < EMB; k0 += 32) {
        FragU a[4], bfr[4];
#pragma unroll
        for (int i = 0; i < 4; ++i) {
            const ushort_t* ap = ab + (size_t)(mrow + i*16 + n16) * EMB + k0 + 8*half;
            a[i].q[0] = *(const v4u*)(ap);
            a[i].q[1] = *(const v4u*)(ap + 16);
        }
#pragma unroll
        for (int j = 0; j < 4; ++j) {
            const ushort_t* bpw = wpT + (size_t)(ncol + j*16 + n16) * EMB + k0 + 8*half;
            bfr[j].q[0] = *(const v4u*)(bpw);
            bfr[j].q[1] = *(const v4u*)(bpw + 16);
        }
#pragma unroll
        for (int i = 0; i < 4; ++i)
#pragma unroll
            for (int j = 0; j < 4; ++j)
                c[i][j] = __builtin_amdgcn_wmma_f32_16x16x32_bf16(
                    false, a[i].v, false, bfr[j].v, (short)0, c[i][j], false, false);
    }

#pragma unroll
    for (int i = 0; i < 4; ++i) {
#pragma unroll
        for (int j = 0; j < 4; ++j) {
            const int col = ncol + j*16 + n16;
            const float bias = bp[col];
#pragma unroll
            for (int r = 0; r < 8; ++r) {
                const int row = mrow + i*16 + 8*half + r;
                out[(size_t)row*EMB + col] = c[i][j][r] + bias;
            }
        }
    }
}

// ---------------- host launcher ----------------
extern "C" void kernel_launch(void* const* d_in, const int* in_sizes, int n_in,
                              void* d_out, int out_size, void* d_ws, size_t ws_size,
                              hipStream_t stream) {
    const float* x  = (const float*)d_in[0];
    const float* Wk = (const float*)d_in[1];
    const float* Wq = (const float*)d_in[2];
    const float* Wv = (const float*)d_in[3];
    const float* Wp = (const float*)d_in[4];
    const float* bp = (const float*)d_in[5];
    float* out = (float*)d_out;

    char* ws = (char*)d_ws;
    const size_t SZ_XB = (size_t)MROWS * EMB * 2;        // 25,165,824
    const size_t SZ_W  = (size_t)4 * EMB * EMB * 2;      //  1,179,648
    const size_t SZ_H  = (size_t)BATCH * HEADS * SEQ * HS * 2; // 25,165,824
    ushort_t* xb  = (ushort_t*)(ws);
    ushort_t* wT  = (ushort_t*)(ws + SZ_XB);
    ushort_t* qb  = (ushort_t*)(ws + SZ_XB + SZ_W);
    ushort_t* kb  = (ushort_t*)(ws + SZ_XB + SZ_W + SZ_H);
    ushort_t* vT  = (ushort_t*)(ws + SZ_XB + SZ_W + 2*SZ_H);
    ushort_t* ab  = (ushort_t*)(ws + SZ_XB + SZ_W + 3*SZ_H);

    k_cvt_x<<<(MROWS*EMB)/256, 256, 0, stream>>>(x, xb);
    k_cvt_w<<<(4*EMB*EMB)/256, 256, 0, stream>>>(Wq, Wk, Wv, Wp, wT);
    k_qkv_gemm<<<dim3(MROWS/128, EMB/128, 3), 128, 0, stream>>>(xb, wT, qb, kb, vT);
    k_attn<<<BATCH*HEADS*(SEQ/16), 32, 0, stream>>>(qb, kb, vT, ab);
    k_proj_gemm<<<dim3(MROWS/128, EMB/128), 128, 0, stream>>>(
        ab, wT + (size_t)3*EMB*EMB, bp, out);
}
